// QuantumKANClassifier_26688926777597
// MI455X (gfx1250) — compile-verified, matrix operand order checked by reading, not run on hardware
//
#include <hip/hip_runtime.h>

typedef __attribute__((ext_vector_type(16))) _Float16 v16h;
typedef __attribute__((ext_vector_type(8)))  _Float16 v8h;
typedef __attribute__((ext_vector_type(8)))  float    v8f;
typedef __attribute__((ext_vector_type(4)))  float    v4f;

#define BSZ   16384
#define FDIM  512
#define DEG   8
#define CDIM  10
#define KBAS  8
#define FCHUNK 64                 // features per chunk
#define KCHUNK (FCHUNK * KBAS)    // 512 flattened K per chunk
#define NCHUNK (FDIM / FCHUNK)    // 8 chunks
#define WAVES  8
#define TPB    256
#define ROWS   16

// Branch-free tanh on the native TRANS pipe (v_exp_f32 + v_rcp_f32).
__device__ __forceinline__ float fast_tanh(float x) {
    const float a = __builtin_fabsf(x);
    const float e = __expf(-2.0f * a);
    const float t = (1.0f - e) * __builtin_amdgcn_rcpf(1.0f + e);
    return __builtin_copysignf(t, x);
}

__device__ __forceinline__ float fast_sigmoid(float x) {
    return __builtin_amdgcn_rcpf(1.0f + __expf(-x));
}

// A-fragment for v_wmma_f32_16x16x32_f16 from [M][K] LDS:
// lane's 16 halves are two contiguous 8-half runs -> 2x ds_load_b128.
__device__ __forceinline__ v16h load_a_frag(const _Float16* p) {
    const v8h lo = *(const v8h*)p;          // K [hi,   hi+8)
    const v8h hh = *(const v8h*)(p + 16);   // K [16+hi,16+hi+8)
    return __builtin_shufflevector(lo, hh, 0, 1, 2, 3, 4, 5, 6, 7,
                                           8, 9, 10, 11, 12, 13, 14, 15);
}

// B-fragment from row-major f32 global: two contiguous 8-float runs
// -> 4x global_load_b128 + cvt to f16.
__device__ __forceinline__ v16h load_b_frag(const float* p, bool valid) {
    const v4f f0 = *(const v4f*)(p);
    const v4f f1 = *(const v4f*)(p + 4);
    const v4f f2 = *(const v4f*)(p + 16);
    const v4f f3 = *(const v4f*)(p + 20);
    v16h b;
    #pragma unroll
    for (int i = 0; i < 4; ++i) {
        b[i]      = (_Float16)f0[i];
        b[4 + i]  = (_Float16)f1[i];
        b[8 + i]  = (_Float16)f2[i];
        b[12 + i] = (_Float16)f3[i];
    }
    if (!valid) {
        v16h z = {};
        b = z;
    }
    return b;
}

__global__ __launch_bounds__(TPB)
void qkan_wmma_kernel(const float* __restrict__ X,
                      const float* __restrict__ lcu_w,
                      const float* __restrict__ alpha,
                      const float* __restrict__ coeff,
                      const float* __restrict__ base,
                      const float* __restrict__ bias,
                      float* __restrict__ out)
{
    // --- block-resident parameters (loaded once) ---
    __shared__ float wlds[FDIM * DEG];        // 16 KB  lcu weights
    __shared__ float alds[FDIM * DEG];        // 16 KB  sum_alpha phases
    __shared__ float invwn[FDIM];             //  2 KB  1/(L1(w)+eps)
    // --- per-chunk tiles, [M][K] so fragments are contiguous ---
    __shared__ __align__(16) _Float16 phi_lds[ROWS * KCHUNK];   // 16 KB
    __shared__ __align__(16) _Float16 silu_lds[ROWS * FCHUNK];  //  2 KB
    // --- cross-wave reduction ---
    __shared__ float red[WAVES * 256];        //  8 KB

    const int tid  = threadIdx.x;
    const int lane = tid & 31;
    // wave id forced uniform -> scalar branches, no exec masking around WMMA
    const int wu   = __builtin_amdgcn_readfirstlane(tid >> 5);
    const int m0   = blockIdx.x * ROWS;

    // Stage lcu weights / phases into LDS; precompute per-feature inv-norm.
    #pragma unroll
    for (int i = tid; i < FDIM * DEG; i += TPB) {
        wlds[i] = lcu_w[i];
        alds[i] = alpha[i];
    }
    #pragma unroll
    for (int f = tid; f < FDIM; f += TPB) {
        float wn = 0.0f;
        #pragma unroll
        for (int d = 0; d < DEG; ++d)
            wn += __builtin_fabsf(lcu_w[f * DEG + d]);
        invwn[f] = __builtin_amdgcn_rcpf(wn + 1e-8f);
    }
    __syncthreads();

    v8f acc = {};                         // 16x16 f32 accumulator (per wave)

    for (int fc = 0; fc < NCHUNK; ++fc) {
        // ---------------- phase 1: elementwise quantum-KAN chain ----------
        #pragma unroll
        for (int j = 0; j < 4; ++j) {
            const int e  = tid + TPB * j;
            const int m  = e >> 6;        // row in tile
            const int fl = e & 63;        // feature within chunk
            const int f  = fc * FCHUNK + fl;

            const float xin = X[(size_t)(m0 + m) * FDIM + f];
            const float x   = 0.99f * fast_tanh(xin);

            // QSVT basis = Chebyshev T1..T8 via recurrence
            float t0 = 1.0f, t1 = x;
            float dot = 0.0f;
            #pragma unroll
            for (int d = 0; d < DEG; ++d) {
                dot += t1 * wlds[f * DEG + d];
                const float t2 = 2.0f * x * t1 - t0;
                t0 = t1; t1 = t2;
            }
            const float lcu = dot * invwn[f];

            // quantum sum block: mean_d sin(lcu * alpha_d)
            float ssum = 0.0f;
            #pragma unroll
            for (int d = 0; d < DEG; ++d)
                ssum += __sinf(lcu * alds[f * DEG + d]);
            const float summed = ssum * (1.0f / DEG);

            // KAN spline basis; pack 8 halves -> one ds_store_b128
            const float s = 0.99f * fast_tanh(summed);
            v8h ph;
            float u0 = 1.0f, u1 = s;
            #pragma unroll
            for (int k = 0; k < KBAS; ++k) {
                ph[k] = (_Float16)u1;
                const float u2 = 2.0f * s * u1 - u0;
                u0 = u1; u1 = u2;
            }
            *(v8h*)&phi_lds[m * KCHUNK + fl * KBAS] = ph;

            silu_lds[m * FCHUNK + fl] =
                (_Float16)(summed * fast_sigmoid(summed));
        }

        // hint-prefetch next chunk's X slice (global_prefetch_b8)
        if (fc + 1 < NCHUNK) {
            const int m  = tid >> 6;
            const int fl = tid & 63;
            __builtin_prefetch(
                &X[(size_t)(m0 + m) * FDIM + (fc + 1) * FCHUNK + fl], 0, 1);
        }
        __syncthreads();

        // ---------------- phase 2: WMMA contractions ----------------------
        // 16-bit A layout: lanes 0-15 -> M=lane, K {0..7,16..23};
        //                  lanes 16-31 -> M=lane-16, K {8..15,24..31}.
        const int mrow = lane & 15;
        const int hi   = (lane >> 4) << 3;       // +8 K offset, upper lanes
        const int n    = lane & 15;              // class column (pad 10->16)
        const int nld  = (n < CDIM) ? n : 0;     // clamped row for loads
        const bool nok = (n < CDIM);

        // phi branch: wave wu owns K-slice [wu*64, wu*64+64) of this chunk
        #pragma unroll
        for (int ko = 0; ko < 64; ko += 32) {
            const int kl = wu * 64 + ko;             // K base within chunk
            const v16h a = load_a_frag(&phi_lds[mrow * KCHUNK + kl + hi]);
            const v16h b = load_b_frag(
                &coeff[nld * (FDIM * KBAS) + fc * KCHUNK + kl + hi], nok);
            acc = __builtin_amdgcn_wmma_f32_16x16x32_f16(
                      false, a, false, b, (short)0, acc, false, false);
        }

        // silu*base branch: wave fc owns this chunk's 64-feature K range
        if (wu == fc) {
            #pragma unroll
            for (int ko = 0; ko < 64; ko += 32) {
                const v16h a =
                    load_a_frag(&silu_lds[mrow * FCHUNK + ko + hi]);
                const v16h b = load_b_frag(
                    &base[nld * FDIM + fc * FCHUNK + ko + hi], nok);
                acc = __builtin_amdgcn_wmma_f32_16x16x32_f16(
                          false, a, false, b, (short)0, acc, false, false);
            }
        }
        __syncthreads();
    }

    // ---------------- cross-wave reduction + bias + store -----------------
    // C/D layout: lane l, vgpr r -> (M = r + 8*(l>>4), N = l&15)
    #pragma unroll
    for (int r = 0; r < 8; ++r)
        red[wu * 256 + r * 32 + lane] = acc[r];
    __syncthreads();

    if (tid < ROWS * CDIM) {
        const int m = tid / CDIM;
        const int n = tid % CDIM;
        float s = bias[n];
        const int idx = (m & 7) * 32 + n + ((m >> 3) << 4);
        #pragma unroll
        for (int ww = 0; ww < WAVES; ++ww)
            s += red[ww * 256 + idx];
        out[(size_t)(m0 + m) * CDIM + n] = s;
    }
}

extern "C" void kernel_launch(void* const* d_in, const int* in_sizes, int n_in,
                              void* d_out, int out_size, void* d_ws, size_t ws_size,
                              hipStream_t stream) {
    const float* X     = (const float*)d_in[0];   // (B, F)
    const float* lcu_w = (const float*)d_in[1];   // (F, DEG)
    const float* alpha = (const float*)d_in[2];   // (F, DEG)
    const float* coeff = (const float*)d_in[3];   // (C, F, KBASIS)
    const float* base  = (const float*)d_in[4];   // (C, F)
    const float* bias  = (const float*)d_in[5];   // (C,)
    float* out = (float*)d_out;                   // (B, C)

    dim3 grid(BSZ / ROWS);   // 1024 workgroups, one 16-row tile each
    dim3 block(TPB);         // 8 wave32 waves
    qkan_wmma_kernel<<<grid, block, 0, stream>>>(X, lcu_w, alpha, coeff,
                                                 base, bias, out);
}
